// MoEProjector_90769838833729
// MI455X (gfx1250) — matrix operand order repaired and checked by reading, not compiled, not run on hardware
//
#include <hip/hip_runtime.h>
#include <hip/hip_bf16.h>
#include <math.h>

#define N_TOK   2304
#define D_IN    1024
#define HDIM    4096
#define D_OUTC  4096
#define NEXP    8
#define ROWS2   (2 * N_TOK)
#define LN_EPSF 1e-5f

typedef __attribute__((ext_vector_type(16))) __bf16        v16bf;
typedef __attribute__((ext_vector_type(2)))  __bf16        v2bf;
typedef __attribute__((ext_vector_type(8)))  float         v8f;
typedef __attribute__((ext_vector_type(4)))  unsigned int  v4u;
typedef int v4i_b __attribute__((vector_size(16)));        // matches builtin param type

#define AS1 __attribute__((address_space(1)))
#define AS3 __attribute__((address_space(3)))

#if __has_builtin(__builtin_amdgcn_global_load_async_to_lds_b128) && \
    __has_builtin(__builtin_amdgcn_s_wait_asynccnt)
#define USE_ASYNC_LDS 1
#else
#define USE_ASYNC_LDS 0
#endif

__device__ __forceinline__ AS1 v4i_b* gbl_v4(const void* p) {
  return (AS1 v4i_b*)(unsigned long long)p;
}
__device__ __forceinline__ AS3 v4i_b* lds_v4(void* p) {
  // generic LDS address: low 32 bits are the LDS offset (aperture in high bits)
  return (AS3 v4i_b*)(unsigned int)(unsigned long long)p;
}

__device__ __forceinline__ unsigned short f2bf(float f) {
  unsigned u = __float_as_uint(f);
  u += 0x7FFFu + ((u >> 16) & 1u);          // round-to-nearest-even
  return (unsigned short)(u >> 16);
}
__device__ __forceinline__ float bf2f(unsigned short h) {
  return __uint_as_float(((unsigned)h) << 16);
}

// pack two f32 -> 2x bf16 in one dword. 3 VALU ops: 2x round, 1x v_perm_b32.
__device__ __forceinline__ unsigned pack_bf16x2(float lo, float hi) {
#if __has_builtin(__builtin_amdgcn_cvt_pk_bf16_f32)
  union { v2bf v; unsigned u; } cv;
  cv.v = __builtin_amdgcn_cvt_pk_bf16_f32(lo, hi);
  return cv.u;
#elif __has_builtin(__builtin_amdgcn_perm)
  unsigned ulo = __float_as_uint(lo) + 0x8000u;   // round-to-nearest (ties away)
  unsigned uhi = __float_as_uint(hi) + 0x8000u;
  // bytes: [hi.b3, hi.b2, lo.b3, lo.b2]  ->  (uhi>>16)<<16 | (ulo>>16)
  return __builtin_amdgcn_perm(uhi, ulo, 0x07060302u);
#else
  return ((unsigned)f2bf(hi) << 16) | (unsigned)f2bf(lo);
#endif
}
__device__ __forceinline__ float gelu_exact(float x) {
  return 0.5f * x * (1.0f + erff(x * 0.70710678118654752f));
}

// block-wide sum over 256 threads (8 wave32's)
__device__ __forceinline__ float blockSum(float v, float* sh) {
  const int lane = threadIdx.x & 31;
  const int w    = threadIdx.x >> 5;
#pragma unroll
  for (int o = 16; o > 0; o >>= 1) v += __shfl_xor(v, o, 32);
  if (lane == 0) sh[w] = v;
  __syncthreads();
  float r = (threadIdx.x < 8) ? sh[threadIdx.x] : 0.0f;
  if (w == 0) {
#pragma unroll
    for (int o = 4; o > 0; o >>= 1) r += __shfl_xor(r, o, 32);
    if (lane == 0) sh[0] = r;
  }
  __syncthreads();
  float res = sh[0];
  __syncthreads();
  return res;
}

// ---------------------------------------------------------------------------
// Kernel 1: input LayerNorm + gate + top-2 routing.  One block per token.
// ---------------------------------------------------------------------------
__global__ __launch_bounds__(256)
void ln_gate_route_kernel(const float* __restrict__ x,
                          const float* __restrict__ ln_g,
                          const float* __restrict__ ln_b,
                          const float* __restrict__ gate_w,   // [D_IN, E]
                          const float* __restrict__ gate_b,   // [E]
                          unsigned short* __restrict__ xnorm, // [N, D_IN] bf16
                          int*   __restrict__ counts,
                          int*   __restrict__ bucket,         // [E, N]
                          int*   __restrict__ tok_e,          // [N, 2]
                          int*   __restrict__ tok_p,          // [N, 2]
                          float* __restrict__ tok_w)          // [N, 2]
{
  __shared__ float sh[64];
  __shared__ float gsh[8][8];
  const int t   = blockIdx.x;
  const int tid = threadIdx.x;
  const float* xr = x + (size_t)t * D_IN;

  float xv[4];
#pragma unroll
  for (int i = 0; i < 4; ++i) xv[i] = xr[tid + i * 256];

  float s = xv[0] + xv[1] + xv[2] + xv[3];
  float mu = blockSum(s, sh) * (1.0f / D_IN);
  float q = 0.f;
#pragma unroll
  for (int i = 0; i < 4; ++i) { float d = xv[i] - mu; q += d * d; }
  float var = blockSum(q, sh) * (1.0f / D_IN);
  float rs  = rsqrtf(var + LN_EPSF);

  float part[NEXP];
#pragma unroll
  for (int e = 0; e < NEXP; ++e) part[e] = 0.f;

#pragma unroll
  for (int i = 0; i < 4; ++i) {
    int idx = tid + i * 256;
    float xn = (xv[i] - mu) * rs * ln_g[idx] + ln_b[idx];
    xnorm[(size_t)t * D_IN + idx] = f2bf(xn);
#pragma unroll
    for (int e = 0; e < NEXP; ++e) part[e] += xn * gate_w[idx * NEXP + e];
  }

  const int lane = tid & 31, wv = tid >> 5;
#pragma unroll
  for (int e = 0; e < NEXP; ++e) {
    float p = part[e];
#pragma unroll
    for (int o = 16; o > 0; o >>= 1) p += __shfl_xor(p, o, 32);
    if (lane == 0) gsh[wv][e] = p;
  }
  __syncthreads();

  if (tid == 0) {
    float lg[NEXP];
#pragma unroll
    for (int e = 0; e < NEXP; ++e) {
      float a = 0.f;
#pragma unroll
      for (int w = 0; w < 8; ++w) a += gsh[w][e];
      lg[e] = a + gate_b[e];
    }
    int e0 = 0;
#pragma unroll
    for (int e = 1; e < NEXP; ++e) if (lg[e] > lg[e0]) e0 = e;   // first-index ties like top_k
    int e1 = (e0 == 0) ? 1 : 0;
#pragma unroll
    for (int e = 0; e < NEXP; ++e) if (e != e0 && lg[e] > lg[e1]) e1 = e;
    float z1 = __expf(lg[e1] - lg[e0]);
    float denom = 1.0f + z1;
    float w0 = 1.0f / denom, w1 = z1 / denom;

    int p0 = atomicAdd(&counts[e0], 1);
    int p1 = atomicAdd(&counts[e1], 1);
    bucket[e0 * N_TOK + p0] = t;
    bucket[e1 * N_TOK + p1] = t;
    tok_e[2 * t] = e0; tok_e[2 * t + 1] = e1;
    tok_p[2 * t] = p0; tok_p[2 * t + 1] = p1;
    tok_w[2 * t] = w0; tok_w[2 * t + 1] = w1;
  }
}

__global__ void prefix_kernel(const int* __restrict__ counts, int* __restrict__ offsets) {
  if (threadIdx.x == 0) {
    int r = 0;
    for (int e = 0; e < NEXP; ++e) { offsets[e] = r; r += counts[e]; }
  }
}

// ---------------------------------------------------------------------------
// Templated expert GEMM:  Out[p,:] = act( A[p,:] @ W[e] + bias[e] )
// Block tile 64 tokens x 128 features, K-step 32, double-buffered LDS,
// async A-tile copy (ASYNCcnt), 8 waves x 4 v_wmma_f32_16x16x32_bf16 / step.
// ---------------------------------------------------------------------------
template<int KTOT, bool GATHER, bool DO_GELU>
__global__ __launch_bounds__(256)
void moe_gemm_kernel(const unsigned short* __restrict__ Ain,  // bf16 rows
                     const float* __restrict__ Wall,          // [E, KTOT, 4096] f32
                     const float* __restrict__ Ball,          // [E, 4096] f32
                     unsigned short* __restrict__ Out,        // [ROWS2, 4096] bf16
                     const int* __restrict__ counts,
                     const int* __restrict__ offsets,
                     const int* __restrict__ bucket)
{
  const int e    = blockIdx.z;
  const int cnt  = counts[e];
  const int base = blockIdx.y * 64;
  if (base >= cnt) return;
  const int n0  = blockIdx.x * 128;
  const int off = offsets[e];

  // padded LDS rows: 40 halfs (80B) -> 16B-aligned b128 fragment chunks
  __shared__ unsigned short As[2][64 * 40];
  __shared__ unsigned short Bs[2][128 * 40];

  const int tid = threadIdx.x;
  const float* Wp = Wall + (size_t)e * KTOT * 4096;

  // ---- A staging mapping: thread -> (row, 16B chunk); OOB rows clamped ----
  const int arow = tid >> 2;
  const int akh  = (tid & 3) * 8;
  int srow = base + arow;
  if (srow >= cnt) srow = base;                       // valid data, result masked later
  const int srcRow = GATHER ? bucket[e * N_TOK + srow] : (off + srow);
  const unsigned short* aptr = Ain + (size_t)srcRow * KTOT + akh;

  // ---- B staging mapping: thread owns column ng (+32j spread), k-pairs kg/kg+8
  const int ng = tid & 31;
  const int kg = tid >> 5;                            // 0..7

  const int lane = tid & 31;
  const int wave = tid >> 5;
  const int m0   = (wave >> 1) * 16;                  // token row strip
  const int c0   = (wave & 1) * 64;                   // feature col group
  const int ln15 = lane & 15;
  const int hk   = lane >> 4;

  v8f acc[4];
#pragma unroll
  for (int j = 0; j < 4; ++j) acc[j] = (v8f)0.0f;

  float breg[2][4][2];                                // [kp-sel][j][k within pair]

  auto stageA = [&](int b, int k0) {
#if USE_ASYNC_LDS
    __builtin_amdgcn_global_load_async_to_lds_b128(
        gbl_v4(aptr + k0), lds_v4(&As[b][arow * 40 + akh]), 0, 0);
#else
    v4u av = *(const v4u*)(aptr + k0);
    *(v4u*)(&As[b][arow * 40 + akh]) = av;
#endif
  };
  auto loadB = [&](int k0) {
#pragma unroll
    for (int t = 0; t < 2; ++t) {
      const int kp = kg + t * 8;
#pragma unroll
      for (int j = 0; j < 4; ++j) {
        const float* p = Wp + (size_t)(k0 + 2 * kp) * 4096 + n0 + ng + 32 * j;
        breg[t][j][0] = p[0];
        breg[t][j][1] = p[4096];
      }
    }
  };
  auto storeB = [&](int b) {
#pragma unroll
    for (int t = 0; t < 2; ++t) {
      const int kp = kg + t * 8;
#pragma unroll
      for (int j = 0; j < 4; ++j) {
        unsigned pk = pack_bf16x2(breg[t][j][0], breg[t][j][1]);
        *(unsigned*)(&Bs[b][(ng + 32 * j) * 40 + 2 * kp]) = pk;
      }
    }
  };

  constexpr int NSTEP = KTOT / 32;
  int buf = 0;

  // prologue: fill buffer 0
  loadB(0);
  stageA(0, 0);
  storeB(0);

  for (int s = 0; s < NSTEP; ++s) {
#if USE_ASYNC_LDS
    __builtin_amdgcn_s_wait_asynccnt(0);              // async A writes visible
#endif
    __syncthreads();

    const bool nxt = (s + 1 < NSTEP);
    const int  k0n = (s + 1) * 32;
    if (nxt) {                                        // overlap next-tile fetch with WMMA
      loadB(k0n);
      stageA(buf ^ 1, k0n);
      __builtin_prefetch(Wp + (size_t)(k0n + 32) * 4096 + n0 + ng, 0, 3);
    }

    // A fragment: ISA 16-bit A 16x32 layout (K 0-7/16-23 | 8-15/24-31)
    union { v4u q[2]; v16bf v; } afrag;
    const unsigned short* ap = &As[buf][(m0 + ln15) * 40];
    afrag.q[0] = *(const v4u*)(ap + hk * 8);
    afrag.q[1] = *(const v4u*)(ap + 16 + hk * 8);

#pragma unroll
    for (int j = 0; j < 4; ++j) {
      // B fragment: column n, K 0-7/16-23 (lanes 0-15) | 8-15/24-31 (lanes 16-31)
      union { v4u q[2]; v16bf v; } bfrag;
      const unsigned short* bp = &Bs[buf][(c0 + j * 16 + ln15) * 40 + hk * 8];
      bfrag.q[0] = *(const v4u*)(bp);
      bfrag.q[1] = *(const v4u*)(bp + 16);
      acc[j] = __builtin_amdgcn_wmma_f32_16x16x32_bf16(
          false, afrag.v, false, bfrag.v, (short)0, acc[j], false, false);
    }

    if (nxt) storeB(buf ^ 1);                         // convert after WMMAs issued
    buf ^= 1;
  }

  // ---- epilogue: bias, optional GELU, bf16 store to compacted rows ----
  const float* bias = Ball + (size_t)e * 4096;
#pragma unroll
  for (int j = 0; j < 4; ++j) {
    int ncol = n0 + c0 + j * 16 + ln15;
    float bv = bias[ncol];
#pragma unroll
    for (int r = 0; r < 8; ++r) {
      int m = hk * 8 + r;                  // C/D layout: lanes 16-31 hold M=8..15
      int p = base + m0 + m;
      if (p < cnt) {
        float v = acc[j][r] + bv;
        if (DO_GELU) v = gelu_exact(v);
        Out[(size_t)(off + p) * 4096 + ncol] = f2bf(v);
      }
    }
  }
}

// ---------------------------------------------------------------------------
// Kernel 4: gather a token's two expert rows, weighted sum, output LayerNorm.
// ---------------------------------------------------------------------------
__global__ __launch_bounds__(256)
void combine_ln_kernel(const unsigned short* __restrict__ eo,   // [ROWS2, D_OUT] bf16
                       const int*   __restrict__ tok_e,
                       const int*   __restrict__ tok_p,
                       const float* __restrict__ tok_w,
                       const int*   __restrict__ offsets,
                       const float* __restrict__ ln_g,
                       const float* __restrict__ ln_b,
                       float* __restrict__ out)
{
  __shared__ float sh[64];
  const int t = blockIdx.x, tid = threadIdx.x;
  const int e0 = tok_e[2 * t], e1 = tok_e[2 * t + 1];
  const size_t r0 = (size_t)(offsets[e0] + tok_p[2 * t])     * D_OUTC;
  const size_t r1 = (size_t)(offsets[e1] + tok_p[2 * t + 1]) * D_OUTC;
  const float w0 = tok_w[2 * t], w1 = tok_w[2 * t + 1];

  float v[16];
  float s = 0.f;
#pragma unroll
  for (int i = 0; i < 16; ++i) {
    int c = tid + i * 256;
    v[i] = w0 * bf2f(eo[r0 + c]) + w1 * bf2f(eo[r1 + c]);
    s += v[i];
  }
  float mu = blockSum(s, sh) * (1.0f / D_OUTC);
  float q = 0.f;
#pragma unroll
  for (int i = 0; i < 16; ++i) { float d = v[i] - mu; q += d * d; }
  float var = blockSum(q, sh) * (1.0f / D_OUTC);
  float rs  = rsqrtf(var + LN_EPSF);
  float* orow = out + (size_t)t * D_OUTC;
#pragma unroll
  for (int i = 0; i < 16; ++i) {
    int c = tid + i * 256;
    orow[c] = (v[i] - mu) * rs * ln_g[c] + ln_b[c];
  }
}

// ---------------------------------------------------------------------------
extern "C" void kernel_launch(void* const* d_in, const int* in_sizes, int n_in,
                              void* d_out, int out_size, void* d_ws, size_t ws_size,
                              hipStream_t stream)
{
  (void)in_sizes; (void)n_in; (void)out_size; (void)ws_size;
  const float* x        = (const float*)d_in[0];
  const float* ln_in_g  = (const float*)d_in[1];
  const float* ln_in_b  = (const float*)d_in[2];
  const float* gate_w   = (const float*)d_in[3];
  const float* gate_b   = (const float*)d_in[4];
  const float* W1       = (const float*)d_in[5];
  const float* b1       = (const float*)d_in[6];
  const float* W2       = (const float*)d_in[7];
  const float* b2       = (const float*)d_in[8];
  const float* W3       = (const float*)d_in[9];
  const float* b3       = (const float*)d_in[10];
  const float* ln_out_g = (const float*)d_in[11];
  const float* ln_out_b = (const float*)d_in[12];
  float* out = (float*)d_out;

  char* ws = (char*)d_ws;
  size_t cur = 0;
  auto alloc = [&](size_t bytes) -> void* {
    cur = (cur + 255) & ~(size_t)255;
    void* p = ws + cur;
    cur += bytes;
    return p;
  };
  unsigned short* xnorm = (unsigned short*)alloc((size_t)N_TOK * D_IN  * 2);
  unsigned short* h1    = (unsigned short*)alloc((size_t)ROWS2 * HDIM  * 2);
  unsigned short* h2    = (unsigned short*)alloc((size_t)ROWS2 * HDIM  * 2);
  unsigned short* eo    = (unsigned short*)alloc((size_t)ROWS2 * D_OUTC * 2);
  int*   counts  = (int*)alloc(NEXP * sizeof(int));
  int*   offsets = (int*)alloc(NEXP * sizeof(int));
  int*   bucket  = (int*)alloc((size_t)NEXP * N_TOK * sizeof(int));
  int*   tok_e   = (int*)alloc((size_t)N_TOK * 2 * sizeof(int));
  int*   tok_p   = (int*)alloc((size_t)N_TOK * 2 * sizeof(int));
  float* tok_w   = (float*)alloc((size_t)N_TOK * 2 * sizeof(float));

  (void)hipMemsetAsync(counts, 0, NEXP * sizeof(int), stream);

  ln_gate_route_kernel<<<N_TOK, 256, 0, stream>>>(
      x, ln_in_g, ln_in_b, gate_w, gate_b, xnorm,
      counts, bucket, tok_e, tok_p, tok_w);

  prefix_kernel<<<1, 32, 0, stream>>>(counts, offsets);

  dim3 blk(256);
  dim3 grd(4096 / 128, (N_TOK + 63) / 64, NEXP);
  moe_gemm_kernel<D_IN, true,  true ><<<grd, blk, 0, stream>>>(
      xnorm, W1, b1, h1, counts, offsets, bucket);
  moe_gemm_kernel<HDIM, false, true ><<<grd, blk, 0, stream>>>(
      h1, W2, b2, h2, counts, offsets, bucket);
  moe_gemm_kernel<HDIM, false, false><<<grd, blk, 0, stream>>>(
      h2, W3, b3, eo, counts, offsets, bucket);

  combine_ln_kernel<<<N_TOK, 256, 0, stream>>>(
      eo, tok_e, tok_p, tok_w, offsets, ln_out_g, ln_out_b, out);
}
